// DecoderRNN_18854906429854
// MI455X (gfx1250) — matrix operand order repaired and checked by reading, not compiled, not run on hardware
//
#include <hip/hip_runtime.h>

// ---------------------------------------------------------------------------
// Show-Attend-Tell decoder for MI455X (gfx1250, wave32, WMMA bf16)
// ---------------------------------------------------------------------------
#define B_   32
#define P_   196
#define ENC_ 2048
#define DEC_ 512
#define ATT_ 512
#define E_   256
#define V_   30000
#define L_   21
#define T_   20
#define XH_  (E_ + ENC_ + DEC_)   // 2816 : [emb | awe | h] fused input row
#define GK_  XH_                  // K for gates GEMM

typedef __attribute__((ext_vector_type(16))) __bf16 bf16x16;
typedef __attribute__((ext_vector_type(8)))  float  f32x8;

// fp32 -> bf16 round-to-nearest-even (stored as ushort)
__device__ __forceinline__ unsigned short f2bf(float x) {
    unsigned int u = __float_as_uint(x);
    unsigned int r = u + 0x7FFFu + ((u >> 16) & 1u);
    return (unsigned short)(r >> 16);
}

__device__ __forceinline__ float sigmoidf(float x) {
    return 1.0f / (1.0f + expf(-x));
}

// ---------------------------------------------------------------------------
// Elementwise fp32 -> bf16
// ---------------------------------------------------------------------------
__global__ __launch_bounds__(256)
void f32_to_bf16_kernel(const float* __restrict__ src,
                        unsigned short* __restrict__ dst, int n) {
    int i = blockIdx.x * 256 + threadIdx.x;
    if (i < n) dst[i] = f2bf(src[i]);
}

// Pack [W_ih | W_hh] -> Wg bf16 [2048, 2816]
__global__ __launch_bounds__(256)
void pack_wg_kernel(const float* __restrict__ W_ih,
                    const float* __restrict__ W_hh,
                    unsigned short* __restrict__ Wg) {
    int i = blockIdx.x * 256 + threadIdx.x;
    if (i >= 4 * DEC_ * GK_) return;
    int g = i / GK_, col = i % GK_;
    float v = (col < (E_ + ENC_)) ? W_ih[g * (E_ + ENC_) + col]
                                  : W_hh[g * DEC_ + (col - (E_ + ENC_))];
    Wg[i] = f2bf(v);
}

// ---------------------------------------------------------------------------
// mean over P:  mean[b,e] = (1/P) sum_p enc[b,p,e]
// ---------------------------------------------------------------------------
__global__ __launch_bounds__(256)
void mean_kernel(const float* __restrict__ enc, float* __restrict__ mean) {
    int i = blockIdx.x * 256 + threadIdx.x;       // over B*ENC
    if (i >= B_ * ENC_) return;
    int b = i / ENC_, e = i % ENC_;
    const float* p = enc + (size_t)b * P_ * ENC_ + e;
    float s = 0.f;
    for (int k = 0; k < P_; ++k) s += p[(size_t)k * ENC_];
    mean[i] = s * (1.0f / P_);
}

// h0/c0 = mean @ W.T + b ; also seeds bf16 h-slot of fused xh row
__global__ __launch_bounds__(256)
void init_hc_kernel(const float* __restrict__ mean,
                    const float* __restrict__ W_h0, const float* __restrict__ b_h0,
                    const float* __restrict__ W_c0, const float* __restrict__ b_c0,
                    float* __restrict__ h, float* __restrict__ c,
                    unsigned short* __restrict__ xh) {
    int j = blockIdx.x * 256 + threadIdx.x;       // 0..2*DEC-1
    int b = blockIdx.y;
    if (j >= 2 * DEC_) return;
    bool is_h = (j < DEC_);
    int d = is_h ? j : j - DEC_;
    const float* W = is_h ? W_h0 : W_c0;
    const float* mb = mean + (size_t)b * ENC_;
    const float* Wr = W + (size_t)d * ENC_;
    float acc = is_h ? b_h0[d] : b_c0[d];
    for (int e = 0; e < ENC_; ++e) acc += mb[e] * Wr[e];
    if (is_h) {
        h[b * DEC_ + d] = acc;
        xh[(size_t)b * XH_ + (E_ + ENC_) + d] = f2bf(acc);
    } else {
        c[b * DEC_ + d] = acc;
    }
}

// ---------------------------------------------------------------------------
// Generic bf16 WMMA GEMM:  C[M,N] = A[M,K] @ W[N,K]^T (+bias[N])
// One wave computes one 16x16 tile.  grid = (ceil(Ntiles/8), Mtiles), blk=256.
// All K's are multiples of 32; all pointers/strides keep 16B alignment.
// ---------------------------------------------------------------------------
__global__ __launch_bounds__(256)
void gemm_bf16_wmma(const unsigned short* __restrict__ A, int lda,
                    const unsigned short* __restrict__ W, int ldw,
                    float* __restrict__ C, int ldc,
                    const float* __restrict__ bias, int Ntiles, int K) {
    int wave = threadIdx.x >> 5;
    int lane = threadIdx.x & 31;
    int tn = blockIdx.x * 8 + wave;
    if (tn >= Ntiles) return;                 // wave-uniform: EXEC stays full
    int m0   = blockIdx.y * 16;
    int n0   = tn * 16;
    int half = lane >> 4;                     // 0: lanes 0-15, 1: lanes 16-31
    int l    = lane & 15;

    const unsigned short* Arow = A + (size_t)(m0 + l) * lda;  // row m = m0+l
    const unsigned short* Wrow = W + (size_t)(n0 + l) * ldw;  // col n = n0+l

    f32x8 acc = {};
    for (int k0 = 0; k0 < K; k0 += 32) {
        // A 16x32 bf16: lane half=0 -> K 0-7 & 16-23 ; half=1 -> K 8-15 & 24-31
        union { uint4 q[2]; bf16x16 v; } ua;
        ua.q[0] = *reinterpret_cast<const uint4*>(Arow + k0 + half * 8);
        ua.q[1] = *reinterpret_cast<const uint4*>(Arow + k0 + 16 + half * 8);
        // B 32x16 bf16 (B[k][n] = W[n][k]): lane half=0 -> K 0-7 & 16-23 ; half=1 -> K 8-15 & 24-31
        union { uint4 q[2]; bf16x16 v; } ub;
        ub.q[0] = *reinterpret_cast<const uint4*>(Wrow + k0 + half * 8);
        ub.q[1] = *reinterpret_cast<const uint4*>(Wrow + k0 + 16 + half * 8);

        acc = __builtin_amdgcn_wmma_f32_16x16x32_bf16(
            false, ua.v, false, ub.v, (short)0, acc, false, false);
    }

    float bv = bias ? bias[n0 + l] : 0.0f;
    // C/D layout: VGPR r -> (M = r + 8*half, N = l)
    float* Cp = C + (size_t)(m0 + half * 8) * ldc + n0 + l;
#pragma unroll
    for (int r = 0; r < 8; ++r)
        Cp[(size_t)r * ldc] = acc[r] + bv;
}

// ---------------------------------------------------------------------------
// att2[b,a] = h[b,:] . Wd_att[a,:] + bd_att[a]
// ---------------------------------------------------------------------------
__global__ __launch_bounds__(256)
void att2_kernel(const float* __restrict__ h,
                 const float* __restrict__ Wd, const float* __restrict__ bd,
                 float* __restrict__ att2) {
    int a = blockIdx.x * 256 + threadIdx.x;
    int b = blockIdx.y;
    if (a >= ATT_) return;
    const float* hb = h + (size_t)b * DEC_;
    const float* Wr = Wd + (size_t)a * DEC_;
    float acc = bd[a];
    for (int d = 0; d < DEC_; ++d) acc += hb[d] * Wr[d];
    att2[b * ATT_ + a] = acc;
}

// ---------------------------------------------------------------------------
// Per-batch attention: e -> softmax(alpha) -> awe ; writes alpha to d_out
// and bf16(awe) into the fused xh row.  One block (256 thr) per b.
// ---------------------------------------------------------------------------
__global__ __launch_bounds__(256)
void attention_kernel(const float* __restrict__ att1,   // [B,P,ATT]
                      const float* __restrict__ att2,   // [B,ATT]
                      const float* __restrict__ Wf,     // [ATT]
                      const float* __restrict__ bf,     // [1]
                      const float* __restrict__ enc,    // [B,P,ENC]
                      float* __restrict__ d_out, int t,
                      unsigned short* __restrict__ xh) {
    __shared__ float s_att2[ATT_];
    __shared__ float s_wf[ATT_];
    __shared__ float s_e[256];
    __shared__ float s_red[256];

    int tid  = threadIdx.x;
    int b    = blockIdx.x;
    int wave = tid >> 5, lane = tid & 31;

    for (int i = tid; i < ATT_; i += 256) {
        s_att2[i] = att2[b * ATT_ + i];
        s_wf[i]   = Wf[i];
    }
    __syncthreads();

    // e[p] = bf + sum_a wf[a]*relu(att1[b,p,a]+att2[b,a]) ; wave per p
    const float* a1b = att1 + (size_t)b * P_ * ATT_;
    for (int p = wave; p < P_; p += 8) {
        const float* row = a1b + (size_t)p * ATT_;
        float acc = 0.f;
        for (int a = lane; a < ATT_; a += 32) {
            float v = row[a] + s_att2[a];
            acc += s_wf[a] * (v > 0.f ? v : 0.f);
        }
#pragma unroll
        for (int off = 16; off > 0; off >>= 1)
            acc += __shfl_xor(acc, off, 32);
        if (lane == 0) s_e[p] = acc + bf[0];
    }
    __syncthreads();

    // softmax over 196
    s_red[tid] = (tid < P_) ? s_e[tid] : -3.4e38f;
    __syncthreads();
    for (int s = 128; s > 0; s >>= 1) {
        if (tid < s) s_red[tid] = fmaxf(s_red[tid], s_red[tid + s]);
        __syncthreads();
    }
    float mx = s_red[0];
    __syncthreads();
    float ev = (tid < P_) ? expf(s_e[tid] - mx) : 0.f;
    s_red[tid] = ev;
    __syncthreads();
    for (int s = 128; s > 0; s >>= 1) {
        if (tid < s) s_red[tid] += s_red[tid + s];
        __syncthreads();
    }
    float inv = 1.0f / s_red[0];
    __syncthreads();
    if (tid < P_) {
        float alpha = ev * inv;
        s_e[tid] = alpha;
        d_out[(size_t)B_ * T_ * V_ + ((size_t)b * T_ + t) * P_ + tid] = alpha;
    }
    __syncthreads();

    // awe[e] = sum_p alpha[p]*enc[b,p,e]  -> bf16 into xh[b, E_ + e]
    const float* eb = enc + (size_t)b * P_ * ENC_;
    for (int e = tid; e < ENC_; e += 256) {
        float acc = 0.f;
        for (int p = 0; p < P_; ++p)
            acc += s_e[p] * eb[(size_t)p * ENC_ + e];
        xh[(size_t)b * XH_ + E_ + e] = f2bf(acc);
    }
}

// embedding lookup -> bf16 into xh[b, 0..E-1]
__global__ __launch_bounds__(256)
void build_x_kernel(const float* __restrict__ emb,
                    const int* __restrict__ captions, int t,
                    unsigned short* __restrict__ xh) {
    int i = blockIdx.x * 256 + threadIdx.x;       // over B*E
    if (i >= B_ * E_) return;
    int b = i / E_, j = i % E_;
    int tok = captions[b * L_ + t];
    xh[(size_t)b * XH_ + j] = f2bf(emb[(size_t)tok * E_ + j]);
}

// LSTM pointwise; updates h,c (fp32) and bf16 h-slot of xh
__global__ __launch_bounds__(256)
void lstm_kernel(const float* __restrict__ gates,     // [B,4*DEC] (no bias yet)
                 const float* __restrict__ b_ih, const float* __restrict__ b_hh,
                 float* __restrict__ h, float* __restrict__ c,
                 unsigned short* __restrict__ xh) {
    int i = blockIdx.x * 256 + threadIdx.x;       // over B*DEC
    if (i >= B_ * DEC_) return;
    int b = i / DEC_, d = i % DEC_;
    const float* g4 = gates + (size_t)b * 4 * DEC_;
    float gi = g4[d]            + b_ih[d]            + b_hh[d];
    float gf = g4[DEC_ + d]     + b_ih[DEC_ + d]     + b_hh[DEC_ + d];
    float gg = g4[2 * DEC_ + d] + b_ih[2 * DEC_ + d] + b_hh[2 * DEC_ + d];
    float go = g4[3 * DEC_ + d] + b_ih[3 * DEC_ + d] + b_hh[3 * DEC_ + d];
    float cn = sigmoidf(gf) * c[i] + sigmoidf(gi) * tanhf(gg);
    float hn = sigmoidf(go) * tanhf(cn);
    c[i] = cn;
    h[i] = hn;
    xh[(size_t)b * XH_ + (E_ + ENC_) + d] = f2bf(hn);
}

// ---------------------------------------------------------------------------
extern "C" void kernel_launch(void* const* d_in, const int* in_sizes, int n_in,
                              void* d_out, int out_size, void* d_ws, size_t ws_size,
                              hipStream_t stream) {
    const float* encoder_out = (const float*)d_in[0];
    const int*   captions    = (const int*)  d_in[1];
    // d_in[2] caption_lengths: unused by reference math
    const float* emb    = (const float*)d_in[3];
    const float* We_att = (const float*)d_in[4];
    const float* be_att = (const float*)d_in[5];
    const float* Wd_att = (const float*)d_in[6];
    const float* bd_att = (const float*)d_in[7];
    const float* Wf_att = (const float*)d_in[8];
    const float* bf_att = (const float*)d_in[9];
    const float* W_ih   = (const float*)d_in[10];
    const float* b_ih   = (const float*)d_in[11];
    const float* W_hh   = (const float*)d_in[12];
    const float* b_hh   = (const float*)d_in[13];
    const float* W_fc   = (const float*)d_in[14];
    const float* b_fc   = (const float*)d_in[15];
    const float* W_h0   = (const float*)d_in[16];
    const float* b_h0   = (const float*)d_in[17];
    const float* W_c0   = (const float*)d_in[18];
    const float* b_c0   = (const float*)d_in[19];
    float* out = (float*)d_out;

    // ---- workspace carve (all 256B aligned) ----
    char* base = (char*)d_ws;
    size_t off = 0;
    auto alloc = [&](size_t bytes) -> void* {
        void* p = base + off;
        off = (off + bytes + 255) & ~(size_t)255;
        return p;
    };
    unsigned short* enc_bf = (unsigned short*)alloc((size_t)B_ * P_ * ENC_ * 2);
    unsigned short* We_bf  = (unsigned short*)alloc((size_t)ATT_ * ENC_ * 2);
    unsigned short* Wfc_bf = (unsigned short*)alloc((size_t)V_ * DEC_ * 2);
    unsigned short* Wg_bf  = (unsigned short*)alloc((size_t)4 * DEC_ * GK_ * 2);
    float* att1  = (float*)alloc((size_t)B_ * P_ * ATT_ * 4);
    float* meanb = (float*)alloc((size_t)B_ * ENC_ * 4);
    float* h     = (float*)alloc((size_t)B_ * DEC_ * 4);
    float* c     = (float*)alloc((size_t)B_ * DEC_ * 4);
    float* att2  = (float*)alloc((size_t)B_ * ATT_ * 4);
    float* gates = (float*)alloc((size_t)B_ * 4 * DEC_ * 4);
    unsigned short* xh = (unsigned short*)alloc((size_t)B_ * XH_ * 2);
    (void)ws_size; (void)n_in; (void)in_sizes; (void)out_size;

    // ---- one-time: bf16 conversions + packing ----
    {
        int n = B_ * P_ * ENC_;
        f32_to_bf16_kernel<<<(n + 255) / 256, 256, 0, stream>>>(encoder_out, enc_bf, n);
        n = ATT_ * ENC_;
        f32_to_bf16_kernel<<<(n + 255) / 256, 256, 0, stream>>>(We_att, We_bf, n);
        n = V_ * DEC_;
        f32_to_bf16_kernel<<<(n + 255) / 256, 256, 0, stream>>>(W_fc, Wfc_bf, n);
        n = 4 * DEC_ * GK_;
        pack_wg_kernel<<<(n + 255) / 256, 256, 0, stream>>>(W_ih, W_hh, Wg_bf);
    }

    // ---- init h,c from mean-pooled encoder ----
    mean_kernel<<<(B_ * ENC_ + 255) / 256, 256, 0, stream>>>(encoder_out, meanb);
    init_hc_kernel<<<dim3(4, B_), 256, 0, stream>>>(meanb, W_h0, b_h0, W_c0, b_c0,
                                                    h, c, xh);

    // ---- att1 = enc @ We_att.T + be_att   (M=6272, N=512, K=2048) ----
    gemm_bf16_wmma<<<dim3((ATT_ / 16 + 7) / 8, (B_ * P_) / 16), 256, 0, stream>>>(
        enc_bf, ENC_, We_bf, ENC_, att1, ATT_, be_att, ATT_ / 16, ENC_);

    // ---- 20 sequential decode steps ----
    for (int t = 0; t < T_; ++t) {
        att2_kernel<<<dim3(ATT_ / 256, B_), 256, 0, stream>>>(h, Wd_att, bd_att, att2);
        attention_kernel<<<B_, 256, 0, stream>>>(att1, att2, Wf_att, bf_att,
                                                 encoder_out, out, t, xh);
        build_x_kernel<<<(B_ * E_ + 255) / 256, 256, 0, stream>>>(emb, captions, t, xh);

        // gates = [x|awe|h] @ [W_ih|W_hh].T   (M=32, N=2048, K=2816)
        gemm_bf16_wmma<<<dim3((4 * DEC_ / 16 + 7) / 8, B_ / 16), 256, 0, stream>>>(
            xh, XH_, Wg_bf, GK_, gates, 4 * DEC_, nullptr, 4 * DEC_ / 16, GK_);

        lstm_kernel<<<(B_ * DEC_ + 255) / 256, 256, 0, stream>>>(gates, b_ih, b_hh,
                                                                 h, c, xh);

        // preds[:, t, :] = h_new @ W_fc.T + b_fc  (M=32, N=30000, K=512)
        // A = h-slot of fused xh row (bf16), lda = 2816; C strided into d_out.
        gemm_bf16_wmma<<<dim3((V_ / 16 + 7) / 8, B_ / 16), 256, 0, stream>>>(
            xh + (E_ + ENC_), XH_, Wfc_bf, DEC_, out + (size_t)t * V_, T_ * V_,
            b_fc, V_ / 16, DEC_);
    }
}
